// AttentionGroupAggregator_15685220565130
// MI455X (gfx1250) — matrix operand order repaired and checked by reading, not compiled, not run on hardware
//
#include <hip/hip_runtime.h>

#define NROWS 262144
#define DDIM 128

typedef __attribute__((ext_vector_type(16))) __bf16 v16bf;
typedef __attribute__((ext_vector_type(8)))  float v8f;

// ---------------------------------------------------------------------------
// K0: zero the float-atomic accumulators in workspace (ws is poisoned 0xAA).
// scal layout: [0]=m (max of s), [1]=S (sum exp), [2]=M2 (max logit), [3]=Z
// ---------------------------------------------------------------------------
__global__ void init_kernel(float* scal, float* sumeE, float* ge) {
  int t = threadIdx.x;
  if (t < DDIM) { sumeE[t] = 0.f; ge[t] = 0.f; }
  if (t == 0) { scal[1] = 0.f; scal[3] = 0.f; }
}

// ---------------------------------------------------------------------------
// K1: per-row dot products sp = E.item, s = E.w_pa + b_pa ; per-block max(s)
// ---------------------------------------------------------------------------
__global__ __launch_bounds__(256) void dots_kernel(
    const float* __restrict__ E, const float* __restrict__ item,
    const float* __restrict__ wpa, const float* __restrict__ bpa,
    float* __restrict__ sp, float* __restrict__ s, float* __restrict__ blockmax)
{
  const int lane = threadIdx.x & 31;
  const int wave = threadIdx.x >> 5;
  const int wid  = blockIdx.x * (blockDim.x >> 5) + wave;
  const int nwaves = (gridDim.x * blockDim.x) >> 5;
  float4 iv = *(const float4*)(item + 4 * lane);
  float4 wv = *(const float4*)(wpa  + 4 * lane);
  const float bp = bpa[0];
  float wmax = -3.4e38f;
  for (int row = wid; row < NROWS; row += nwaves) {
    float4 ev = *(const float4*)(E + (size_t)row * DDIM + 4 * lane);
    float dsp = ev.x * iv.x + ev.y * iv.y + ev.z * iv.z + ev.w * iv.w;
    float dss = ev.x * wv.x + ev.y * wv.y + ev.z * wv.z + ev.w * wv.w;
#pragma unroll
    for (int msk = 1; msk < 32; msk <<= 1) {
      dsp += __shfl_xor(dsp, msk, 32);
      dss += __shfl_xor(dss, msk, 32);
    }
    dss += bp;
    if (lane == 0) { sp[row] = dsp; s[row] = dss; }
    wmax = fmaxf(wmax, dss);
  }
  __shared__ float sm[8];
  if (lane == 0) sm[wave] = wmax;
  __syncthreads();
  if (threadIdx.x == 0) {
    float m = sm[0];
    for (int i = 1; i < (int)(blockDim.x >> 5); ++i) m = fmaxf(m, sm[i]);
    blockmax[blockIdx.x] = m;
  }
}

// ---------------------------------------------------------------------------
// generic max reductions
// ---------------------------------------------------------------------------
__global__ __launch_bounds__(256) void block_max_kernel(
    const float* __restrict__ src, int n, float* __restrict__ part)
{
  __shared__ float sm[256];
  float m = -3.4e38f;
  for (int i = blockIdx.x * blockDim.x + threadIdx.x; i < n; i += gridDim.x * blockDim.x)
    m = fmaxf(m, src[i]);
  sm[threadIdx.x] = m;
  __syncthreads();
  for (int g = 128; g > 0; g >>= 1) {
    if ((int)threadIdx.x < g) sm[threadIdx.x] = fmaxf(sm[threadIdx.x], sm[threadIdx.x + g]);
    __syncthreads();
  }
  if (threadIdx.x == 0) part[blockIdx.x] = sm[0];
}

__global__ __launch_bounds__(256) void reduce_max_final(
    const float* __restrict__ part, int n, float* __restrict__ out)
{
  __shared__ float sm[256];
  float m = -3.4e38f;
  for (int i = threadIdx.x; i < n; i += blockDim.x) m = fmaxf(m, part[i]);
  sm[threadIdx.x] = m;
  __syncthreads();
  for (int g = 128; g > 0; g >>= 1) {
    if ((int)threadIdx.x < g) sm[threadIdx.x] = fmaxf(sm[threadIdx.x], sm[threadIdx.x + g]);
    __syncthreads();
  }
  if (threadIdx.x == 0) *out = sm[0];
}

// ---------------------------------------------------------------------------
// K3: e_i = exp(s_i - m); accumulate S and sum_eE[128]
// ---------------------------------------------------------------------------
__global__ __launch_bounds__(256) void exp_sum_kernel(
    const float* __restrict__ E, const float* __restrict__ s,
    const float* __restrict__ scal,
    float* __restrict__ e, float* __restrict__ Sout, float* __restrict__ sumeE)
{
  const int lane = threadIdx.x & 31;
  const int wave = threadIdx.x >> 5;
  const int wid  = blockIdx.x * (blockDim.x >> 5) + wave;
  const int nwaves = (gridDim.x * blockDim.x) >> 5;
  const float m = scal[0];
  float4 acc = make_float4(0.f, 0.f, 0.f, 0.f);
  float Sloc = 0.f;
  for (int row = wid; row < NROWS; row += nwaves) {
    float ei = expf(s[row] - m);
    float4 ev = *(const float4*)(E + (size_t)row * DDIM + 4 * lane);
    acc.x += ei * ev.x; acc.y += ei * ev.y; acc.z += ei * ev.z; acc.w += ei * ev.w;
    if (lane == 0) { e[row] = ei; Sloc += ei; }
  }
  __shared__ float smS[8];
  __shared__ float smV[8 * DDIM];
  *(float4*)(smV + wave * DDIM + 4 * lane) = acc;
  if (lane == 0) smS[wave] = Sloc;
  __syncthreads();
  if (threadIdx.x < DDIM) {
    float t = 0.f;
    for (int w = 0; w < 8; ++w) t += smV[w * DDIM + threadIdx.x];
    atomicAdd(&sumeE[threadIdx.x], t);
  }
  if (threadIdx.x == 0) {
    float t = 0.f;
    for (int w = 0; w < 8; ++w) t += smS[w];
    atomicAdd(Sout, t);
  }
}

// ---------------------------------------------------------------------------
// K4: c[j] = sum_eE @ W2^T ; bsum = b1+b2 ; convert W1,W2 to bf16 (row-major)
// ---------------------------------------------------------------------------
__global__ void prep_kernel(
    const float* __restrict__ W1, const float* __restrict__ W2,
    const float* __restrict__ b1, const float* __restrict__ b2,
    const float* __restrict__ sumeE,
    float* __restrict__ cvec, float* __restrict__ bsumv,
    __bf16* __restrict__ W1bf, __bf16* __restrict__ W2bf)
{
  const int j = threadIdx.x; // 128 threads
  float cj = 0.f;
  for (int k = 0; k < DDIM; ++k) cj += sumeE[k] * W2[j * DDIM + k];
  cvec[j] = cj;
  bsumv[j] = b1[j] + b2[j];
  for (int k = 0; k < DDIM; ++k) {
    W1bf[j * DDIM + k] = (__bf16)W1[j * DDIM + k];
    W2bf[j * DDIM + k] = (__bf16)W2[j * DDIM + k];
  }
}

// ---------------------------------------------------------------------------
// K5: WMMA main pass. W1/W2 (bf16) + per-column scalars staged in LDS once
// per block; each wave grid-strides over 16-row tiles, computing both GEMMs
// (shared A operand), then the fused epilogue:
//   h = g1 + bsum + (c - e_i*g2)/denom_i ; pi = relu(h).w_vc ;
//   logit = sp + pi + b_vc.
// A layout (16-bit A 16x32): lanes 0-15 row M=l, K=0..15 (contiguous);
// lanes 16-31 row M=l-16, K=16..31.
// B layout (32x16): lane holds col N=l&15, K=(l>>4)*16 .. +15 (contig in W row).
// C/D: VGPR r, lane l -> row M=r+8*(l>>4), col N=l&15.
// ---------------------------------------------------------------------------
__global__ __launch_bounds__(256) void gemm_pi_kernel(
    const float* __restrict__ E,
    const __bf16* __restrict__ W1bf, const __bf16* __restrict__ W2bf,
    const float* __restrict__ sp, const float* __restrict__ e,
    const float* __restrict__ cvec, const float* __restrict__ bsumv,
    const float* __restrict__ wvc, const float* __restrict__ scal,
    const float* __restrict__ bvc, float* __restrict__ logit)
{
  __shared__ __align__(32) __bf16 sW1[DDIM * DDIM];
  __shared__ __align__(32) __bf16 sW2[DDIM * DDIM];
  __shared__ float sC[DDIM];
  __shared__ float sB[DDIM];
  __shared__ float sWv[DDIM];

  // Stage both weight matrices (32 KB each) + per-column scalars into LDS.
  {
    const uint4* g1 = (const uint4*)W1bf;
    const uint4* g2 = (const uint4*)W2bf;
    uint4* t1 = (uint4*)sW1;
    uint4* t2 = (uint4*)sW2;
    for (int idx = threadIdx.x; idx < (DDIM * DDIM) / 8; idx += 256) {
      t1[idx] = g1[idx];
      t2[idx] = g2[idx];
    }
    if (threadIdx.x < DDIM) {
      sC[threadIdx.x]  = cvec[threadIdx.x];
      sB[threadIdx.x]  = bsumv[threadIdx.x];
      sWv[threadIdx.x] = wvc[threadIdx.x];
    }
  }

  const int lane = threadIdx.x & 31;
  const int wave = threadIdx.x >> 5;
  const int half = lane >> 4;
  const int l15  = lane & 15;

  const float S  = scal[1];
  const float bv = bvc[0];

  __syncthreads();

  const int wave0  = blockIdx.x * 8 + wave;
  const int nwaves = gridDim.x * 8;

  for (int tile = wave0; tile < NROWS / 16; tile += nwaves) {
    const int i0 = tile * 16;
    const int arow  = i0 + l15;
    const int rbase = i0 + half * 8;

    float er[8], inv[8];
    {
      float4 e0 = *(const float4*)(e + rbase);
      float4 e1 = *(const float4*)(e + rbase + 4);
      er[0] = e0.x; er[1] = e0.y; er[2] = e0.z; er[3] = e0.w;
      er[4] = e1.x; er[5] = e1.y; er[6] = e1.z; er[7] = e1.w;
#pragma unroll
      for (int r = 0; r < 8; ++r) inv[r] = __builtin_amdgcn_rcpf(S - er[r]);
    }

    // Load E rows once, convert f32 -> bf16 into the 4 K-chunks of A.
    v16bf A[4];
    {
      const float* rp = E + (size_t)arow * DDIM;
#pragma unroll
      for (int kc = 0; kc < 4; ++kc) {
        const int cb = kc * 32 + half * 16;
        float4 f0 = *(const float4*)(rp + cb);
        float4 f1 = *(const float4*)(rp + cb + 4);
        float4 f2 = *(const float4*)(rp + cb + 8);
        float4 f3 = *(const float4*)(rp + cb + 12);
        v16bf a;
        a[0]  = (__bf16)f0.x; a[1]  = (__bf16)f0.y; a[2]  = (__bf16)f0.z; a[3]  = (__bf16)f0.w;
        a[4]  = (__bf16)f1.x; a[5]  = (__bf16)f1.y; a[6]  = (__bf16)f1.z; a[7]  = (__bf16)f1.w;
        a[8]  = (__bf16)f2.x; a[9]  = (__bf16)f2.y; a[10] = (__bf16)f2.z; a[11] = (__bf16)f2.w;
        a[12] = (__bf16)f3.x; a[13] = (__bf16)f3.y; a[14] = (__bf16)f3.z; a[15] = (__bf16)f3.w;
        A[kc] = a;
      }
    }

    float piacc[8];
#pragma unroll
    for (int r = 0; r < 8; ++r) piacc[r] = 0.f;

    // Partial unroll: enough cross-iteration overlap for ds_load/WMMA
    // pipelining, small enough live window to avoid scratch spills.
#pragma unroll 2
    for (int jt = 0; jt < 8; ++jt) {
      const int j = jt * 16 + l15;
      v8f acc1 = {};
      v8f acc2 = {};
      const __bf16* b1p = sW1 + j * DDIM + half * 16;
      const __bf16* b2p = sW2 + j * DDIM + half * 16;
#pragma unroll
      for (int kc = 0; kc < 4; ++kc) {
        v16bf B1 = *(const v16bf*)(b1p + kc * 32);
        v16bf B2 = *(const v16bf*)(b2p + kc * 32);
        acc1 = __builtin_amdgcn_wmma_f32_16x16x32_bf16(false, A[kc], false, B1,
                                                       (short)0, acc1, false, false);
        acc2 = __builtin_amdgcn_wmma_f32_16x16x32_bf16(false, A[kc], false, B2,
                                                       (short)0, acc2, false, false);
      }
      // Per-column scalars from LDS (computed address, no register indexing).
      const float cj = sC[j];
      const float bj = sB[j];
      const float wj = sWv[j];
#pragma unroll
      for (int r = 0; r < 8; ++r) {
        float h = acc1[r] + bj + (cj - er[r] * acc2[r]) * inv[r];
        float rh = h > 0.f ? h : 0.f; // ReLU without canonicalize
        piacc[r] += rh * wj;
      }
    }

    // reduce over the 16 lanes sharing each row (xor within lane bits 0..3)
#pragma unroll
    for (int r = 0; r < 8; ++r) {
      float v = piacc[r];
      v += __shfl_xor(v, 1, 32);
      v += __shfl_xor(v, 2, 32);
      v += __shfl_xor(v, 4, 32);
      v += __shfl_xor(v, 8, 32);
      piacc[r] = v;
    }

    if (l15 == 0) { // lanes 0 and 16 write 8 rows each
#pragma unroll
      for (int r = 0; r < 8; ++r) {
        const int i = rbase + r;
        logit[i] = sp[i] + piacc[r] + bv;
      }
    }
  }
}

// ---------------------------------------------------------------------------
// K8: ew = exp(logit - M2); accumulate Z
// ---------------------------------------------------------------------------
__global__ __launch_bounds__(256) void expz_kernel(
    const float* __restrict__ logit, const float* __restrict__ scal, int n,
    float* __restrict__ ew, float* __restrict__ Z)
{
  __shared__ float sm[256];
  const float M2 = scal[2];
  float t = 0.f;
  for (int i = blockIdx.x * blockDim.x + threadIdx.x; i < n; i += gridDim.x * blockDim.x) {
    float v = expf(logit[i] - M2);
    ew[i] = v;
    t += v;
  }
  sm[threadIdx.x] = t;
  __syncthreads();
  for (int g = 128; g > 0; g >>= 1) {
    if ((int)threadIdx.x < g) sm[threadIdx.x] += sm[threadIdx.x + g];
    __syncthreads();
  }
  if (threadIdx.x == 0) atomicAdd(Z, sm[0]);
}

// ---------------------------------------------------------------------------
// K9: attention weights to d_out[128..], accumulate group embedding
// ---------------------------------------------------------------------------
__global__ __launch_bounds__(256) void out_kernel(
    const float* __restrict__ E, const float* __restrict__ ew,
    const float* __restrict__ scal, float* __restrict__ out, float* __restrict__ ge)
{
  const int lane = threadIdx.x & 31;
  const int wave = threadIdx.x >> 5;
  const int wid  = blockIdx.x * (blockDim.x >> 5) + wave;
  const int nwaves = (gridDim.x * blockDim.x) >> 5;
  const float invZ = 1.0f / scal[3];
  float4 acc = make_float4(0.f, 0.f, 0.f, 0.f);
  for (int row = wid; row < NROWS; row += nwaves) {
    float wi = ew[row] * invZ;
    float4 ev = *(const float4*)(E + (size_t)row * DDIM + 4 * lane);
    acc.x += wi * ev.x; acc.y += wi * ev.y; acc.z += wi * ev.z; acc.w += wi * ev.w;
    if (lane == 0) out[DDIM + row] = wi;
  }
  __shared__ float smV[8 * DDIM];
  *(float4*)(smV + wave * DDIM + 4 * lane) = acc;
  __syncthreads();
  if (threadIdx.x < DDIM) {
    float t = 0.f;
    for (int w = 0; w < 8; ++w) t += smV[w * DDIM + threadIdx.x];
    atomicAdd(&ge[threadIdx.x], t);
  }
}

__global__ void final_ge_kernel(const float* __restrict__ ge, float* __restrict__ out) {
  out[threadIdx.x] = ge[threadIdx.x]; // 128 threads; ew already normalized by invZ
}

// ---------------------------------------------------------------------------
extern "C" void kernel_launch(void* const* d_in, const int* in_sizes, int n_in,
                              void* d_out, int out_size, void* d_ws, size_t ws_size,
                              hipStream_t stream) {
  const float* E    = (const float*)d_in[0];
  const float* item = (const float*)d_in[1];
  const float* W1   = (const float*)d_in[2];
  const float* b1   = (const float*)d_in[3];
  const float* W2   = (const float*)d_in[4];
  const float* b2   = (const float*)d_in[5];
  const float* wpa  = (const float*)d_in[6];
  const float* bpa  = (const float*)d_in[7];
  const float* wvc  = (const float*)d_in[8];
  const float* bvc  = (const float*)d_in[9];
  float* out = (float*)d_out;

  float* ws = (float*)d_ws;
  const size_t Nn = NROWS;
  float* sp    = ws;
  float* s     = ws + Nn;
  float* e     = ws + 2 * Nn;
  float* logit = ws + 3 * Nn;
  float* ew    = ws + 4 * Nn;
  float* part  = ws + 5 * Nn;   // up to 32768 partials
  float* scal  = part + 32768;  // [m, S, M2, Z]
  float* sumeE = scal + 4;
  float* cvec  = sumeE + DDIM;
  float* bsumv = cvec + DDIM;
  float* ge    = bsumv + DDIM;
  size_t off = (size_t)((ge + DDIM) - ws);
  off = (off + 7) & ~(size_t)7; // 32-byte align for bf16 vector loads
  __bf16* W1bf = (__bf16*)(ws + off);
  __bf16* W2bf = W1bf + DDIM * DDIM;

  hipLaunchKernelGGL(init_kernel, dim3(1), dim3(256), 0, stream, scal, sumeE, ge);
  hipLaunchKernelGGL(dots_kernel, dim3(4096), dim3(256), 0, stream,
                     E, item, wpa, bpa, sp, s, part);
  hipLaunchKernelGGL(reduce_max_final, dim3(1), dim3(256), 0, stream, part, 4096, scal + 0);
  hipLaunchKernelGGL(exp_sum_kernel, dim3(1024), dim3(256), 0, stream,
                     E, s, scal, e, scal + 1, sumeE);
  hipLaunchKernelGGL(prep_kernel, dim3(1), dim3(128), 0, stream,
                     W1, W2, b1, b2, sumeE, cvec, bsumv, W1bf, W2bf);
  hipLaunchKernelGGL(gemm_pi_kernel, dim3(512), dim3(256), 0, stream,
                     E, W1bf, W2bf, sp, e, cvec, bsumv, wvc, scal, bvc, logit);
  hipLaunchKernelGGL(block_max_kernel, dim3(1024), dim3(256), 0, stream,
                     logit, NROWS, part);
  hipLaunchKernelGGL(reduce_max_final, dim3(1), dim3(256), 0, stream, part, 1024, scal + 2);
  hipLaunchKernelGGL(expz_kernel, dim3(1024), dim3(256), 0, stream,
                     logit, scal, NROWS, ew, scal + 3);
  hipLaunchKernelGGL(out_kernel, dim3(1024), dim3(256), 0, stream, E, ew, scal, out, ge);
  hipLaunchKernelGGL(final_ge_kernel, dim3(1), dim3(128), 0, stream, ge, out);
}
